// DynamicConvBlock_65730179498326
// MI455X (gfx1250) — matrix operand order repaired and checked by reading, not compiled, not run on hardware
//
#include <hip/hip_runtime.h>
#include <hip/hip_bf16.h>

// ---------------------------------------------------------------------------
// DynamicConvBlock fused for gfx1250 (MI455X).
// Math identity: einsum("btkd,btkf->btf") = sum_k rowsum(x)[t+k-3] * kernels[t,k,f]
// => two WMMA GEMMs + cheap tap-combine, fused LN/relu epilogue.
// ---------------------------------------------------------------------------

#define BB 8
#define TT 2048
#define DD 512
#define FF 512
#define KK 7
#define HH 128

typedef __attribute__((ext_vector_type(16))) _Float16 v16h;
typedef __attribute__((ext_vector_type(8)))  _Float16 v8h;
typedef __attribute__((ext_vector_type(8)))  float    v8f;

__device__ __forceinline__ v16h cat16(v8h lo, v8h hi) {
  return __builtin_shufflevector(lo, hi, 0,1,2,3,4,5,6,7,8,9,10,11,12,13,14,15);
}

// --- Kernel 0: transpose + fp32->fp16 for B operands ------------------------
// W1T[h][d] = W1[d][h]   (128 x 512 f16)
// W2T[c][h] = W2[h][c]   (3584 x 128 f16)
__global__ void dcb_prep(const float* __restrict__ W1, const float* __restrict__ W2,
                         _Float16* __restrict__ W1T, _Float16* __restrict__ W2T) {
  int id = blockIdx.x * 256 + threadIdx.x;
  if (id < DD * HH) {
    int h = id >> 9, d = id & (DD - 1);
    W1T[id] = (_Float16)W1[d * HH + h];
  }
  int id2 = id - DD * HH;
  if (id2 >= 0 && id2 < KK * FF * HH) {
    int c = id2 >> 7, hh = id2 & (HH - 1);
    W2T[id2] = (_Float16)W2[hh * (KK * FF) + c];
  }
}

// --- Kernel 1: per-token row sums s[b*T+t] = sum_d x ------------------------
__global__ void dcb_rowsum(const float* __restrict__ x, float* __restrict__ s) {
  int t = blockIdx.x * 256 + threadIdx.x;   // 0..16383
  const float4* p = (const float4*)(x + (size_t)t * DD);
  float acc = 0.f;
#pragma unroll 8
  for (int i = 0; i < DD / 4; ++i) { float4 v = p[i]; acc += v.x + v.y + v.z + v.w; }
  s[t] = acc;
}

// --- Kernel 2: H = relu(x @ W1 + b1), stored f16 [16384][128] ---------------
// 32-token tile per WG, 8 waves. wave: mt = w&1 (16-row M-tile),
// two h-tiles (w>>1)*2 .. +1. K loop over D in steps of 32.
__global__ __launch_bounds__(256) void dcb_gemm1(const float* __restrict__ x,
                                                 const _Float16* __restrict__ W1T,
                                                 const float* __restrict__ b1g,
                                                 _Float16* __restrict__ Hout) {
  int t0   = blockIdx.x * 32;
  int lane = threadIdx.x & 31, w = threadIdx.x >> 5;
  int mt = w & 1, ht0 = (w >> 1) * 2;
  int hi = (lane >> 4) & 1, ln = lane & 15;
  const float* xrow = x + (size_t)(t0 + mt * 16 + ln) * DD;

  v8f acc0 = {}; v8f acc1 = {};
#pragma unroll
  for (int i = 0; i < DD / 32; ++i) {
    // A fragment: 16-bit A layout -> halves j<8: d=32i+8*hi+j ; j>=8: +16
    int d0 = 32 * i + 8 * hi;
    float fl[16];
    *(float4*)(fl + 0)  = *(const float4*)(xrow + d0 + 0);
    *(float4*)(fl + 4)  = *(const float4*)(xrow + d0 + 4);
    *(float4*)(fl + 8)  = *(const float4*)(xrow + d0 + 16);
    *(float4*)(fl + 12) = *(const float4*)(xrow + d0 + 20);
    v16h a;
#pragma unroll
    for (int j = 0; j < 16; ++j) a[j] = (_Float16)fl[j];

    // B fragments: column h in lane, K runs at 8*hi and 8*hi+16 (same as A)
    const v8h* bp0 = (const v8h*)(W1T + (size_t)(ht0 * 16 + ln) * DD + 32 * i + 8 * hi);
    const v8h* bp1 = (const v8h*)(W1T + (size_t)((ht0 + 1) * 16 + ln) * DD + 32 * i + 8 * hi);
    v16h bf0 = cat16(bp0[0], bp0[2]);
    v16h bf1 = cat16(bp1[0], bp1[2]);
    acc0 = __builtin_amdgcn_wmma_f32_16x16x32_f16(false, a, false, bf0, (short)0, acc0, false, false);
    acc1 = __builtin_amdgcn_wmma_f32_16x16x32_f16(false, a, false, bf1, (short)0, acc1, false, false);
  }
#pragma unroll
  for (int tt = 0; tt < 2; ++tt) {
    int h = (ht0 + tt) * 16 + ln;
    float bb = b1g[h];
#pragma unroll
    for (int r = 0; r < 8; ++r) {
      int row = t0 + mt * 16 + r + 8 * hi;           // C/D layout
      float v = (tt ? acc1[r] : acc0[r]) + bb;
      Hout[(size_t)row * HH + h] = (_Float16)(v > 0.f ? v : 0.f);
    }
  }
}

// --- Kernel 3: fused GEMM2 + tap-combine + residual + LayerNorm + relu ------
__global__ __launch_bounds__(256) void dcb_gemm2(const float* __restrict__ x,
                                                 const _Float16* __restrict__ Hm,
                                                 const _Float16* __restrict__ W2T,
                                                 const float* __restrict__ s,
                                                 const float* __restrict__ b2,
                                                 const float* __restrict__ gamma,
                                                 const float* __restrict__ beta,
                                                 float* __restrict__ out) {
  __shared__ float y_lds[32 * FF];                   // 64 KB
  int t0   = blockIdx.x * 32;
  int tb   = t0 & (TT - 1);                          // offset within batch
  int lane = threadIdx.x & 31, w = threadIdx.x >> 5;
  int mt = w & 1, fgrp = w >> 1;                     // fgrp 0..3 -> 128 f each
  int hi = (lane >> 4) & 1, ln = lane & 15;

  // cached A fragments of H (f16), K = 0..127
  int trow = t0 + mt * 16 + ln;
  v16h afr[4];
#pragma unroll
  for (int i = 0; i < 4; ++i) {
    const v8h* p = (const v8h*)(Hm + (size_t)trow * HH + 32 * i + 8 * hi);
    afr[i] = cat16(p[0], p[2]);                      // runs at +0 and +16 halves
  }

  // tap weights per output row: wk[k][r] = s[t + k - 3] (zero at batch edges)
  float wk[KK][8];
#pragma unroll
  for (int k = 0; k < KK; ++k)
#pragma unroll
    for (int r = 0; r < 8; ++r) {
      int m  = mt * 16 + r + 8 * hi;
      int ts = tb + m + k - 3;
      wk[k][r] = (ts >= 0 && ts < TT) ? s[(t0 - tb) + ts] : 0.f;
    }

#pragma unroll 1
  for (int ftl = 0; ftl < 8; ++ftl) {
    int f0  = (fgrp * 8 + ftl) * 16;
    int col = f0 + ln;
    float accf[8];
#pragma unroll
    for (int r = 0; r < 8; ++r) accf[r] = 0.f;

#pragma unroll
    for (int k = 0; k < KK; ++k) {
      v8f tmp = {};
#pragma unroll
      for (int i = 0; i < 4; ++i) {
        const v8h* bp = (const v8h*)(W2T + (size_t)(k * FF + col) * HH + 32 * i + 8 * hi);
        v16h bfr = cat16(bp[0], bp[2]);
        tmp = __builtin_amdgcn_wmma_f32_16x16x32_f16(false, afr[i], false, bfr, (short)0, tmp, false, false);
      }
      float b2v = b2[k * FF + col];
#pragma unroll
      for (int r = 0; r < 8; ++r) accf[r] += wk[k][r] * (tmp[r] + b2v);
    }
    // residual, stage to LDS for LayerNorm
#pragma unroll
    for (int r = 0; r < 8; ++r) {
      int m = mt * 16 + r + 8 * hi;
      y_lds[m * FF + col] = accf[r] + x[(size_t)(t0 + m) * DD + col];
    }
  }
  __syncthreads();

  // LayerNorm + relu: wave w owns tokens 4w..4w+3; 8 lanes per token.
  int tok = w * 4 + (lane >> 3);
  int sub = lane & 7;
  const float4* yp = (const float4*)(y_lds + tok * FF + sub * 64);
  float sum = 0.f, sq = 0.f;
#pragma unroll
  for (int i = 0; i < 16; ++i) {
    float4 v = yp[i];
    sum += v.x + v.y + v.z + v.w;
    sq  += v.x * v.x + v.y * v.y + v.z * v.z + v.w * v.w;
  }
#pragma unroll
  for (int off = 1; off < 8; off <<= 1) {
    sum += __shfl_xor(sum, off, 32);
    sq  += __shfl_xor(sq,  off, 32);
  }
  float mu  = sum * (1.f / FF);
  float var = sq * (1.f / FF) - mu * mu;
  float rs  = rsqrtf(var + 1e-3f);
  const float4* gp = (const float4*)(gamma + sub * 64);
  const float4* bp = (const float4*)(beta  + sub * 64);
  float4* op = (float4*)(out + (size_t)(t0 + tok) * FF + sub * 64);
#pragma unroll
  for (int i = 0; i < 16; ++i) {
    float4 v = yp[i], g = gp[i], be = bp[i], o;
    o.x = fmaxf((v.x - mu) * rs * g.x + be.x, 0.f);
    o.y = fmaxf((v.y - mu) * rs * g.y + be.y, 0.f);
    o.z = fmaxf((v.z - mu) * rs * g.z + be.z, 0.f);
    o.w = fmaxf((v.w - mu) * rs * g.w + be.w, 0.f);
    op[i] = o;
  }
}

extern "C" void kernel_launch(void* const* d_in, const int* in_sizes, int n_in,
                              void* d_out, int out_size, void* d_ws, size_t ws_size,
                              hipStream_t stream) {
  const float* x     = (const float*)d_in[0];
  const float* W1    = (const float*)d_in[1];
  const float* b1    = (const float*)d_in[2];
  const float* W2    = (const float*)d_in[3];
  const float* b2    = (const float*)d_in[4];
  const float* gamma = (const float*)d_in[5];
  const float* beta  = (const float*)d_in[6];
  float* out = (float*)d_out;

  // workspace carve-out (~5.1 MB total)
  char* ws = (char*)d_ws;
  _Float16* W1T = (_Float16*)(ws + 0);          //  512*128*2 = 131072
  _Float16* W2T = (_Float16*)(ws + 131072);     // 3584*128*2 = 917504
  _Float16* Hm  = (_Float16*)(ws + 1048576);    // 16384*128*2 = 4 MB
  float*    sws = (float*)  (ws + 5242880);     // 16384*4

  dcb_prep  <<<(DD * HH + KK * FF * HH) / 256, 256, 0, stream>>>(W1, W2, W1T, W2T);
  dcb_rowsum<<<(BB * TT) / 256,               256, 0, stream>>>(x, sws);
  dcb_gemm1 <<<(BB * TT) / 32,                256, 0, stream>>>(x, W1T, b1, Hm);
  dcb_gemm2 <<<(BB * TT) / 32,                256, 0, stream>>>(x, Hm, W2T, sws, b2, gamma, beta, out);
}
